// MultiheadAttention_37847251813111
// MI455X (gfx1250) — matrix operand order repaired and checked by reading, not compiled, not run on hardware
//
#include <hip/hip_runtime.h>
#include <hip/hip_bf16.h>

// ---------------------------------------------------------------------------
// Enformer-style relative-position MHA for MI455X (gfx1250, wave32, WMMA).
// All heavy math through v_wmma_f32_16x16x32_f16 (f16 in, f32 accumulate).
// relative_shift is fused: rel[q,k] = (q_rr[q] . r_k[k-q+T-1]) computed as a
// 16x32 banded WMMA product + cross-lane diagonal extraction.
// ---------------------------------------------------------------------------

typedef __attribute__((ext_vector_type(16))) _Float16 v16h;
typedef __attribute__((ext_vector_type(8)))  float    v8f;

constexpr int B_  = 2;
constexpr int T_  = 1536;
constexpr int C_  = 1536;
constexpr int H_  = 8;
constexpr int K_  = 64;
constexpr int V_  = 192;
constexpr int HK_ = H_ * K_;    // 512
constexpr int HV_ = H_ * V_;    // 1536
constexpr int BT_ = B_ * T_;    // 3072
constexpr int R_  = 2 * T_ - 1; // 3071
constexpr int RP_ = 3072;       // padded rows for r_k (row 3071 = zeros)

union F16x16 { v16h v; unsigned int u[8]; };

__device__ inline v8f wmma32(v16h a, v16h b, v8f c) {
  return __builtin_amdgcn_wmma_f32_16x16x32_f16(
      false, a, false, b, (short)0, c, false, false);
}

// A fragment, 16x32 f16 (M x K). lane: m = lane&15, half = lane>>4.
// VGPR v holds K pair at (v>>2)*16 + half*8 + (v&3)*2.
// Per-lane bytes are contiguous in two 16B chunks -> lowers to 2x b128.
__device__ inline v16h load_a16x32(const _Float16* base, int ld, int row0,
                                   int kc, int lane) {
  const int m = lane & 15, half = lane >> 4;
  const _Float16* rp = base + (size_t)(row0 + m) * ld + kc;
  F16x16 f;
#pragma unroll
  for (int v = 0; v < 8; ++v) {
    int kb = ((v >> 2) << 4) + (half << 3) + ((v & 3) << 1);
    f.u[v] = *(const unsigned int*)(rp + kb);
  }
  return f.v;
}

// B fragment, 32x16 f16 (K x N), with operand stored N-major: elem(kk,n) at
// base[(n0+n)*ld + kc + kk]. lane: n = lane&15, K pair = (v>>2)*16 + (lane>>4)*8 + (v&3)*2.
// Per-lane bytes contiguous in two 16B chunks -> lowers to 2x b128.
__device__ inline v16h load_b32x16(const _Float16* base, int ld, int n0,
                                   int kc, int lane) {
  const int n = lane & 15, half = lane >> 4;
  const _Float16* rp = base + (size_t)(n0 + n) * ld + kc;
  F16x16 f;
#pragma unroll
  for (int v = 0; v < 8; ++v)
    f.u[v] = *(const unsigned int*)(rp + ((v >> 2) << 4) + (half << 3) + ((v & 3) << 1));
  return f.v;
}

// ---------------------------------------------------------------------------
// Converts
// ---------------------------------------------------------------------------
__global__ void cvt_f16_kernel(const float* __restrict__ s,
                               _Float16* __restrict__ d, long n) {
  long i = (long)blockIdx.x * blockDim.x + threadIdx.x;
  if (i < n) d[i] = (_Float16)s[i];
}

// dst[c*rows + r] = src[r*cols + c]  (store weight transposed: [N][Cin])
__global__ void cvt_t_kernel(const float* __restrict__ s,
                             _Float16* __restrict__ d, int rows, int cols) {
  long i = (long)blockIdx.x * blockDim.x + threadIdx.x;
  long n = (long)rows * cols;
  if (i >= n) return;
  int r = (int)(i / cols), c = (int)(i - (long)r * cols);
  d[(size_t)c * rows + r] = (_Float16)s[i];
}

// ---------------------------------------------------------------------------
// r_k = position_features @ Wrk, stored f16 [H][RP_][64]; pad row zeroed.
// feats: 32 central-mask indicators + sign(d) * same.
// ---------------------------------------------------------------------------
__global__ void rk_kernel(const float* __restrict__ Wrk,
                          _Float16* __restrict__ rk16) {
  long i = (long)blockIdx.x * blockDim.x + threadIdx.x;
  if (i >= (long)RP_ * HK_) return;
  int r = (int)(i >> 9), col = (int)(i & 511);
  float acc = 0.0f;
  if (r < R_) {
    float d  = (float)(r - (T_ - 1));
    float sg = (d > 0.0f) ? 1.0f : (d < 0.0f ? -1.0f : 0.0f);
    float ad = fabsf(d);
    float rate = __expf(__logf((float)(T_ + 1)) / 32.0f);
    float w = rate;
#pragma unroll
    for (int j = 0; j < 32; ++j) {
      if ((w - 1.0f) > ad)
        acc += Wrk[j * HK_ + col] + sg * Wrk[(32 + j) * HK_ + col];
      w *= rate;
    }
  }
  int h = col >> 6, kk = col & 63;
  rk16[((size_t)h * RP_ + r) * K_ + kk] = (_Float16)acc;
}

// ---------------------------------------------------------------------------
// Generic WMMA GEMM: C[M,N] = A[M,Cin] (row-major f16) x Wt[N][Cin] (f16).
// Block: 256 thr = 8 waves; block tile 256(M) x 64(N); wave tile 32 x 64
// (two A row-fragments reuse each B fragment -> 8 WMMAs per 6 frag loads).
// K-loop is a 2-deep ping-pong: two independent register sets alternate
// consume/refill roles, so no cross-set register copies are needed.
// Requires Cin % 64 == 0 (Cin = 1536 here).
// Epilogue modes:
//   0: k-proj    -> out0 f16 [B,H,T,64]
//   1: q-proj    -> out0 = v*scale+bias0, out1 = v*scale+bias1, [B,H,T,64]
//   2: v-proj    -> out0 f16 [B,H,V,T]  (transposed for flash B-fragments)
//   3: final     -> outf f32 [M,N] + bias0[col]
// ---------------------------------------------------------------------------
__global__ __launch_bounds__(256) void gemm_kernel(
    const _Float16* __restrict__ A, const _Float16* __restrict__ Wt,
    int M, int Cin, int N, int mode, float scale,
    const float* __restrict__ bias0, const float* __restrict__ bias1,
    _Float16* __restrict__ out0, _Float16* __restrict__ out1,
    float* __restrict__ outf) {
  const int tid = threadIdx.x, wave = tid >> 5, lane = tid & 31;
  const int m = lane & 15, half = lane >> 4;
  const int row0 = blockIdx.y * 256 + wave * 32;
  const int col0 = blockIdx.x * 64;

  v8f acc[2][4];
#pragma unroll
  for (int r = 0; r < 2; ++r)
#pragma unroll
    for (int s = 0; s < 4; ++s) acc[r][s] = (v8f){0, 0, 0, 0, 0, 0, 0, 0};

  // prologue: fill both ping-pong sets (kc = 0 and kc = 32)
  v16h aA[2], aB[2], bA[4], bB[4];
#pragma unroll
  for (int r = 0; r < 2; ++r) {
    aA[r] = load_a16x32(A, Cin, row0 + r * 16, 0, lane);
    aB[r] = load_a16x32(A, Cin, row0 + r * 16, 32, lane);
  }
#pragma unroll
  for (int s = 0; s < 4; ++s) {
    bA[s] = load_b32x16(Wt, Cin, col0 + s * 16, 0, lane);
    bB[s] = load_b32x16(Wt, Cin, col0 + s * 16, 32, lane);
  }

  int kc = 0;
  for (; kc < Cin - 64; kc += 64) {
    // consume set A (depth kc), refill it for depth kc+64
#pragma unroll
    for (int r = 0; r < 2; ++r)
#pragma unroll
      for (int s = 0; s < 4; ++s) acc[r][s] = wmma32(aA[r], bA[s], acc[r][s]);
#pragma unroll
    for (int r = 0; r < 2; ++r)
      aA[r] = load_a16x32(A, Cin, row0 + r * 16, kc + 64, lane);
#pragma unroll
    for (int s = 0; s < 4; ++s)
      bA[s] = load_b32x16(Wt, Cin, col0 + s * 16, kc + 64, lane);
    // consume set B (depth kc+32), refill it for depth kc+96
#pragma unroll
    for (int r = 0; r < 2; ++r)
#pragma unroll
      for (int s = 0; s < 4; ++s) acc[r][s] = wmma32(aB[r], bB[s], acc[r][s]);
#pragma unroll
    for (int r = 0; r < 2; ++r)
      aB[r] = load_a16x32(A, Cin, row0 + r * 16, kc + 96, lane);
#pragma unroll
    for (int s = 0; s < 4; ++s)
      bB[s] = load_b32x16(Wt, Cin, col0 + s * 16, kc + 96, lane);
  }
  // tail: consume both sets (depths Cin-64, Cin-32)
#pragma unroll
  for (int r = 0; r < 2; ++r)
#pragma unroll
    for (int s = 0; s < 4; ++s) acc[r][s] = wmma32(aA[r], bA[s], acc[r][s]);
#pragma unroll
  for (int r = 0; r < 2; ++r)
#pragma unroll
    for (int s = 0; s < 4; ++s) acc[r][s] = wmma32(aB[r], bB[s], acc[r][s]);

#pragma unroll
  for (int r = 0; r < 2; ++r) {
#pragma unroll
    for (int s = 0; s < 4; ++s) {
#pragma unroll
      for (int v = 0; v < 8; ++v) {
        int row = row0 + r * 16 + v + (half << 3);
        int col = col0 + s * 16 + m;
        float val = acc[r][s][v];
        if (mode == 3) {
          outf[(size_t)row * N + col] = val + bias0[col];
        } else {
          int bb = row / T_, t = row - bb * T_;
          if (mode == 2) {
            int hh = col / V_, vv = col - hh * V_;
            // transposed: [B,H,V,T]
            out0[((size_t)(bb * H_ + hh) * V_ + vv) * T_ + t] = (_Float16)val;
          } else {
            int hh = col >> 6, kk = col & 63;
            size_t idx = (((size_t)bb * H_ + hh) * T_ + t) * K_ + kk;
            if (mode == 0) {
              out0[idx] = (_Float16)val;
            } else {  // mode 1
              out0[idx] = (_Float16)(val * scale + bias0[col]);
              out1[idx] = (_Float16)(val * scale + bias1[col]);
            }
          }
        }
      }
    }
  }
}

// ---------------------------------------------------------------------------
// Flash attention with fused relative-shift.
// Grid: (T/128, B*H). Block: 256 thr = 8 waves; wave owns 16 query rows.
// k loop in chunks of 32 keys; K tile + (pre-transposed) V tile in LDS,
// staged with b128 copies; next-chunk global_prefetch_b8 issued early.
// ---------------------------------------------------------------------------
__global__ __launch_bounds__(256) void flash_kernel(
    const _Float16* __restrict__ qrw, const _Float16* __restrict__ qrr,
    const _Float16* __restrict__ kmat, const _Float16* __restrict__ vmatT,
    const _Float16* __restrict__ rk16, _Float16* __restrict__ att) {
  __shared__ __align__(16) _Float16 Kl[32 * 64];    // [key][kdim]
  __shared__ __align__(16) _Float16 Vl[192 * 32];   // [vdim][key]
  __shared__ __align__(16) _Float16 Pl[8][16 * 32]; // per-wave P staging

  const int tid = threadIdx.x, wave = tid >> 5, lane = tid & 31;
  const int m = lane & 15, half = lane >> 4;
  const int bh = blockIdx.y, h = bh & 7, b = bh >> 3;
  const int q0 = blockIdx.x * 128 + wave * 16;

  const _Float16* qrwp = qrw + (size_t)bh * T_ * K_;
  const _Float16* qrrp = qrr + (size_t)bh * T_ * K_;
  const _Float16* kp   = kmat + (size_t)bh * T_ * K_;
  const _Float16* vpt  = vmatT + (size_t)bh * V_ * T_;  // [V][T]
  const _Float16* rkh  = rk16 + (size_t)h * RP_ * K_;

  v16h qa0 = load_a16x32(qrwp, K_, q0, 0, lane);
  v16h qa1 = load_a16x32(qrwp, K_, q0, 32, lane);
  v16h qr0 = load_a16x32(qrrp, K_, q0, 0, lane);
  v16h qr1 = load_a16x32(qrrp, K_, q0, 32, lane);

  float mrow[8], lrow[8];
  v8f oacc[12];
#pragma unroll
  for (int v = 0; v < 8; ++v) { mrow[v] = -3.0e30f; lrow[v] = 0.0f; }
#pragma unroll
  for (int j = 0; j < 12; ++j) oacc[j] = (v8f){0, 0, 0, 0, 0, 0, 0, 0};

  for (int kb = 0; kb < T_; kb += 32) {
    __syncthreads();
    // Stage K chunk: 32x64 f16 = 4KB, one 16B vector per thread.
    ((uint4*)Kl)[tid] = ((const uint4*)(kp + (size_t)kb * K_))[tid];
    // Stage V chunk: [vdim 192][key 32] = 12KB as 768 x 16B (3 per thread).
    {
      const _Float16* vs = vpt + kb;
#pragma unroll
      for (int it = 0; it < 3; ++it) {
        int i = tid + it * 256;  // 0..767
        int vv = i >> 2, part = i & 3;
        ((uint4*)Vl)[i] = *(const uint4*)(vs + (size_t)vv * T_ + part * 8);
      }
    }
    // Prefetch next chunk into cache while this one is consumed.
    if (kb + 32 < T_) {
      __builtin_prefetch(kp + (size_t)(kb + 32) * K_ + tid * 8, 0, 1);
      if (tid < 192)
        __builtin_prefetch(vpt + (size_t)tid * T_ + kb + 32, 0, 1);
    }
    __syncthreads();

    float larr[2][8];
#pragma unroll
    for (int sub = 0; sub < 2; ++sub) {
      // content logits: Qrw (16x64) x K_sub^T (64x16)
      v8f cacc = (v8f){0, 0, 0, 0, 0, 0, 0, 0};
      cacc = wmma32(qa0, load_b32x16(Kl, K_, sub * 16, 0, lane), cacc);
      cacc = wmma32(qa1, load_b32x16(Kl, K_, sub * 16, 32, lane), cacc);

      // rel logits: window product M = Qrr x rk[o-15 .. o+16]^T (16x32)
      const int o = kb + sub * 16 - q0 + (T_ - 1);
      v8f ma = (v8f){0, 0, 0, 0, 0, 0, 0, 0};
      v8f mb = (v8f){0, 0, 0, 0, 0, 0, 0, 0};
      ma = wmma32(qr0, load_b32x16(rkh, K_, o - 15, 0, lane), ma);
      ma = wmma32(qr1, load_b32x16(rkh, K_, o - 15, 32, lane), ma);
      mb = wmma32(qr0, load_b32x16(rkh, K_, o + 1, 0, lane), mb);
      mb = wmma32(qr1, load_b32x16(rkh, K_, o + 1, 32, lane), mb);

      // diagonal extraction: rel[mm][n] = M[mm][15 + n - mm]
#pragma unroll
      for (int v = 0; v < 8; ++v) {
        int mm = v + (half << 3);
        int col = 15 + m - mm;  // in [0,30]
        float a0 = __shfl(ma[v], (half << 4) | (col & 15), 32);
        float a1 = __shfl(mb[v], (half << 4) | ((col - 16) & 15), 32);
        larr[sub][v] = cacc[v] + ((col < 16) ? a0 : a1);
      }
    }

    // Online softmax over the 32-key chunk (row = 16-lane group).
    float pexp[2][8];
#pragma unroll
    for (int v = 0; v < 8; ++v) {
      float mx = fmaxf(larr[0][v], larr[1][v]);
      mx = fmaxf(mx, __shfl_xor(mx, 1, 32));
      mx = fmaxf(mx, __shfl_xor(mx, 2, 32));
      mx = fmaxf(mx, __shfl_xor(mx, 4, 32));
      mx = fmaxf(mx, __shfl_xor(mx, 8, 32));
      float mnew = fmaxf(mrow[v], mx);
      float alpha = __expf(mrow[v] - mnew);
      mrow[v] = mnew;
      float p0 = __expf(larr[0][v] - mnew);
      float p1 = __expf(larr[1][v] - mnew);
      pexp[0][v] = p0; pexp[1][v] = p1;
      float rs = p0 + p1;
      rs += __shfl_xor(rs, 1, 32);
      rs += __shfl_xor(rs, 2, 32);
      rs += __shfl_xor(rs, 4, 32);
      rs += __shfl_xor(rs, 8, 32);
      lrow[v] = lrow[v] * alpha + rs;
#pragma unroll
      for (int j = 0; j < 12; ++j) oacc[j][v] *= alpha;
    }

    // C-layout -> A-layout for P via per-wave LDS staging (f16).
    _Float16* pw = &Pl[wave][0];
#pragma unroll
    for (int sub = 0; sub < 2; ++sub)
#pragma unroll
      for (int v = 0; v < 8; ++v)
        pw[(v + (half << 3)) * 32 + sub * 16 + m] = (_Float16)pexp[sub][v];
    asm volatile("s_wait_dscnt 0x0" ::: "memory");
    v16h pa;
    {
      F16x16 f;
#pragma unroll
      for (int v = 0; v < 8; ++v) {
        int kbp = ((v >> 2) << 4) + (half << 3) + ((v & 3) << 1);
        f.u[v] = *(const unsigned int*)(pw + m * 32 + kbp);
      }
      pa = f.v;
    }

    // O += P (16x32) x V_chunk (32x192): 12 WMMAs.
#pragma unroll
    for (int j = 0; j < 12; ++j)
      oacc[j] = wmma32(pa, load_b32x16(Vl, 32, j * 16, 0, lane), oacc[j]);
  }

  // Normalize and store to att [B,T,H*V] (f16 for final GEMM).
#pragma unroll
  for (int v = 0; v < 8; ++v) {
    float inv = 1.0f / lrow[v];
    int t = q0 + v + (half << 3);
    _Float16* op = att + ((size_t)b * T_ + t) * HV_ + h * V_;
#pragma unroll
    for (int j = 0; j < 12; ++j)
      op[j * 16 + m] = (_Float16)(oacc[j][v] * inv);
  }
}

// ---------------------------------------------------------------------------
// Host launcher
// ---------------------------------------------------------------------------
extern "C" void kernel_launch(void* const* d_in, const int* in_sizes, int n_in,
                              void* d_out, int out_size, void* d_ws,
                              size_t ws_size, hipStream_t stream) {
  const float* x   = (const float*)d_in[0];
  const float* Wq  = (const float*)d_in[1];
  const float* Wk  = (const float*)d_in[2];
  const float* Wv  = (const float*)d_in[3];
  const float* Wrk = (const float*)d_in[4];
  const float* rwb = (const float*)d_in[5];
  const float* rrb = (const float*)d_in[6];
  const float* Wo  = (const float*)d_in[7];
  const float* bo  = (const float*)d_in[8];
  float* out = (float*)d_out;

  char* p = (char*)d_ws;
  auto alloc = [&](size_t elems) -> _Float16* {
    _Float16* r = (_Float16*)p;
    p += (elems * sizeof(_Float16) + 255) & ~(size_t)255;
    return r;
  };
  _Float16* x16  = alloc((size_t)BT_ * C_);
  _Float16* WqT  = alloc((size_t)C_ * HK_);
  _Float16* WkT  = alloc((size_t)C_ * HK_);
  _Float16* WvT  = alloc((size_t)C_ * HV_);
  _Float16* WoT  = alloc((size_t)HV_ * HV_);
  _Float16* qrw  = alloc((size_t)B_ * H_ * T_ * K_);
  _Float16* qrr  = alloc((size_t)B_ * H_ * T_ * K_);
  _Float16* k16  = alloc((size_t)B_ * H_ * T_ * K_);
  _Float16* v16t = alloc((size_t)B_ * H_ * V_ * T_);
  _Float16* rk16 = alloc((size_t)H_ * RP_ * K_);
  _Float16* att  = alloc((size_t)BT_ * HV_);

  const int TPB = 256;
  // 1) converts (weights transposed to [N][Cin] for contiguous B K-pairs)
  {
    long n = (long)BT_ * C_;
    cvt_f16_kernel<<<dim3((unsigned)((n + TPB - 1) / TPB)), TPB, 0, stream>>>(
        x, x16, n);
  }
  {
    long n = (long)C_ * HK_;
    unsigned g = (unsigned)((n + TPB - 1) / TPB);
    cvt_t_kernel<<<dim3(g), TPB, 0, stream>>>(Wq, WqT, C_, HK_);
    cvt_t_kernel<<<dim3(g), TPB, 0, stream>>>(Wk, WkT, C_, HK_);
  }
  {
    long n = (long)C_ * HV_;
    unsigned g = (unsigned)((n + TPB - 1) / TPB);
    cvt_t_kernel<<<dim3(g), TPB, 0, stream>>>(Wv, WvT, C_, HV_);
    cvt_t_kernel<<<dim3(g), TPB, 0, stream>>>(Wo, WoT, HV_, HV_);
  }
  // 2) r_k table
  {
    long n = (long)RP_ * HK_;
    rk_kernel<<<dim3((unsigned)((n + TPB - 1) / TPB)), TPB, 0, stream>>>(Wrk,
                                                                        rk16);
  }
  // 3) projections (block tile 256 rows x 64 cols)
  const float scale = 0.125f;  // K^-0.5, K=64
  gemm_kernel<<<dim3(HK_ / 64, BT_ / 256), TPB, 0, stream>>>(
      x16, WqT, BT_, C_, HK_, /*mode=*/1, scale, rwb, rrb, qrw, qrr, nullptr);
  gemm_kernel<<<dim3(HK_ / 64, BT_ / 256), TPB, 0, stream>>>(
      x16, WkT, BT_, C_, HK_, /*mode=*/0, 1.0f, nullptr, nullptr, k16, nullptr,
      nullptr);
  gemm_kernel<<<dim3(HV_ / 64, BT_ / 256), TPB, 0, stream>>>(
      x16, WvT, BT_, C_, HV_, /*mode=*/2, 1.0f, nullptr, nullptr, v16t, nullptr,
      nullptr);
  // 4) fused attention (content + shifted rel + softmax + P.V)
  flash_kernel<<<dim3(T_ / 128, B_ * H_), TPB, 0, stream>>>(qrw, qrr, k16, v16t,
                                                            rk16, att);
  // 5) output projection + bias -> f32 d_out [B*T, H*V]
  gemm_kernel<<<dim3(HV_ / 64, BT_ / 256), TPB, 0, stream>>>(
      att, WoT, BT_, C_, HV_, /*mode=*/3, 1.0f, bo, nullptr, nullptr, nullptr,
      out);
  (void)in_sizes; (void)n_in; (void)out_size; (void)ws_size;
}
